// RotaryMultiheadAttention_80796924772526
// MI455X (gfx1250) — matrix operand order repaired and checked by reading, not compiled, not run on hardware
//
#include <hip/hip_runtime.h>
#include <hip/hip_bf16.h>

// ---------------------------------------------------------------------------
// Rotary multi-head attention for MI455X (gfx1250), wave32 + WMMA bf16.
// B=2 S=2048 E=1024 H=16 D=64. Compute-bound (~69 GFLOP vs ~tens of MB of
// traffic against 23.3 TB/s HBM / 192MB L2) -> everything on v_wmma_f32_16x16x32_bf16.
// Attention stages shared K/V chunks into LDS with double-buffered async
// global->LDS copies (GLOBAL_LOAD_ASYNC_TO_LDS_B128, ASYNCcnt-tracked).
// ---------------------------------------------------------------------------

#define B_ 2
#define S_ 2048
#define E_ 1024
#define H_ 16
#define D_ 64
#define NT (B_ * S_)  // 4096 tokens

typedef __bf16 bf16;
typedef __attribute__((ext_vector_type(16))) __bf16 v16bf;
typedef __attribute__((ext_vector_type(8)))  float  v8f;

union V16 { v16bf v; uint4 u[2]; };

#if defined(__has_builtin)
#  if __has_builtin(__builtin_amdgcn_global_load_async_to_lds_b128)
#    define HAVE_ASYNC_COPY 1
#  endif
#  if __has_builtin(__builtin_amdgcn_s_wait_asynccnt)
#    define HAVE_WAIT_ASYNC 1
#  endif
#endif

// Builtin signature (from hipcc diagnostic): (v4i AS1*, v4i AS3*, imm offset, imm cpol)
typedef __attribute__((__vector_size__(16))) int v4i_t;
typedef __attribute__((address_space(1))) v4i_t* gv4i_ptr;
typedef __attribute__((address_space(3))) v4i_t* lv4i_ptr;

__device__ __forceinline__ void async_copy16(const bf16* g, bf16* l) {
#if defined(HAVE_ASYNC_COPY)
  // GLOBAL_LOAD_ASYNC_TO_LDS_B128: per-lane 16B memory->LDS, tracked by ASYNCcnt
  __builtin_amdgcn_global_load_async_to_lds_b128(
      (gv4i_ptr)(void*)g, (lv4i_ptr)l, /*offset=*/0, /*cpol=*/0);
#else
  *(uint4*)l = *(const uint4*)g;  // sync fallback: load + ds_store
#endif
}

__device__ __forceinline__ void async_wait_all() {
#if defined(HAVE_ASYNC_COPY)
#  if defined(HAVE_WAIT_ASYNC)
  __builtin_amdgcn_s_wait_asynccnt(0);
#  else
  asm volatile("s_wait_asynccnt 0" ::: "memory");
#  endif
#endif
}

__device__ __forceinline__ v8f wmma_bf16(v16bf a, v16bf b, v8f c) {
  // D = A(16x32) * B(32x16) + C, f32 accumulate
  return __builtin_amdgcn_wmma_f32_16x16x32_bf16(
      /*neg_a=*/false, a, /*neg_b=*/false, b,
      /*c_mod=*/(short)0, c, /*reuse_a=*/false, /*reuse_b=*/false);
}

// ------------------------------ converters ---------------------------------

__global__ void cvt_x_kernel(const float* __restrict__ x, bf16* __restrict__ xb, int n) {
  int i = blockIdx.x * blockDim.x + threadIdx.x;
  if (i < n) xb[i] = (bf16)x[i];
}

// WT[n][k] = W[k][n]  (pre-transpose so the WMMA B-operand gather is contiguous)
__global__ void cvt_wT_kernel(const float* __restrict__ W, bf16* __restrict__ WT) {
  int i = blockIdx.x * blockDim.x + threadIdx.x;  // over E*E
  int nIdx = i >> 10;
  int k    = i & (E_ - 1);
  WT[i] = (bf16)W[k * E_ + nIdx];
}

// ------------------------- fused QKV projection + RoPE ---------------------
// grid (NT/128, H, 3)  block 256 (8 waves). Wave computes 16 tokens x 64 cols
// (one full head) so the RoPE pair (j, j+32) lives in the same lane.

__global__ __launch_bounds__(256) void qkv_gemm_kernel(
    const bf16* __restrict__ xb,
    const bf16* __restrict__ WqT, const bf16* __restrict__ WkT, const bf16* __restrict__ WvT,
    const float* __restrict__ bq, const float* __restrict__ bk, const float* __restrict__ bv,
    bf16* __restrict__ Qb,  // [B][H][S][D]
    bf16* __restrict__ Kb,  // [B][H][S][D]
    bf16* __restrict__ Vt)  // [B][H][D][S] (transposed for PV B-operand)
{
  const int lane = threadIdx.x & 31;
  const int wv   = threadIdx.x >> 5;
  const int l15  = lane & 15;
  const int half = lane >> 4;
  const int m0   = blockIdx.x * 128 + wv * 16;  // token tile base
  const int h    = blockIdx.y;
  const int n0   = h * D_;
  const int z    = blockIdx.z;

  const bf16*  WT   = (z == 0) ? WqT : ((z == 1) ? WkT : WvT);
  const float* bias = (z == 0) ? bq  : ((z == 1) ? bk  : bv);

  v8f acc0 = {}, acc1 = {}, acc2 = {}, acc3 = {};

  const bf16* Abase = xb + (size_t)(m0 + l15) * E_;   // A row: token m = lane&15
  const bf16* B0    = WT + (size_t)(n0 + l15)      * E_;
  const bf16* B1    = WT + (size_t)(n0 + 16 + l15) * E_;
  const bf16* B2    = WT + (size_t)(n0 + 32 + l15) * E_;
  const bf16* B3    = WT + (size_t)(n0 + 48 + l15) * E_;

  for (int kk = 0; kk < E_; kk += 32) {
    V16 a;
    const bf16* ap = Abase + kk + 16 * half;          // A: k = e + 16*half
    a.u[0] = *(const uint4*)(ap);
    a.u[1] = *(const uint4*)(ap + 8);

    V16 b0, b1, b2, b3;
    const int bo = kk + 16 * half;                    // B: k = e + 16*half
    b0.u[0] = *(const uint4*)(B0 + bo); b0.u[1] = *(const uint4*)(B0 + bo + 8);
    b1.u[0] = *(const uint4*)(B1 + bo); b1.u[1] = *(const uint4*)(B1 + bo + 8);
    b2.u[0] = *(const uint4*)(B2 + bo); b2.u[1] = *(const uint4*)(B2 + bo + 8);
    b3.u[0] = *(const uint4*)(B3 + bo); b3.u[1] = *(const uint4*)(B3 + bo + 8);

    acc0 = wmma_bf16(a.v, b0.v, acc0);
    acc1 = wmma_bf16(a.v, b1.v, acc1);
    acc2 = wmma_bf16(a.v, b2.v, acc2);
    acc3 = wmma_bf16(a.v, b3.v, acc3);
  }

  const float bias0 = bias[n0 + l15];
  const float bias1 = bias[n0 + 16 + l15];
  const float bias2 = bias[n0 + 32 + l15];
  const float bias3 = bias[n0 + 48 + l15];

  // inv_freq(i) = 10000^(-i/32) = exp2(-log2(10000)/32 * i)
  const float FC  = 0.41524101186092034f;
  const float if0 = exp2f(-FC * (float)l15);          // pair index for j0=l15, j2=l15+32
  const float if1 = exp2f(-FC * (float)(l15 + 16));   // pair index for j1, j3

  bf16* QKout = (z == 0) ? Qb : Kb;

#pragma unroll
  for (int r = 0; r < 8; ++r) {
    const int m     = r + 8 * half;       // C/D layout: row = r + 8*(lane>=16)
    const int token = m0 + m;
    const int b     = token >> 11;        // /S_
    const int s     = token & (S_ - 1);
    const int bh    = b * H_ + h;

    float q0 = acc0[r] + bias0;
    float q1 = acc1[r] + bias1;
    float q2 = acc2[r] + bias2;
    float q3 = acc3[r] + bias3;

    if (z < 2) {
      // reference: emb = concat(sin(s*f), cos(s*f)); cosE=cos(emb), sinE=sin(emb)
      const float th0 = (float)s * if0;
      const float th1 = (float)s * if1;
      const float e0 = __sinf(th0), e1 = __sinf(th1);
      const float e2 = __cosf(th0), e3 = __cosf(th1);
      const float o0 = q0 * __cosf(e0) - q2 * __sinf(e0);
      const float o1 = q1 * __cosf(e1) - q3 * __sinf(e1);
      const float o2 = q2 * __cosf(e2) + q0 * __sinf(e2);
      const float o3 = q3 * __cosf(e3) + q1 * __sinf(e3);
      bf16* row = QKout + ((size_t)bh * S_ + s) * D_;
      row[l15]      = (bf16)o0;
      row[l15 + 16] = (bf16)o1;
      row[l15 + 32] = (bf16)o2;
      row[l15 + 48] = (bf16)o3;
    } else {
      bf16* col = Vt + (size_t)bh * D_ * S_ + s;      // Vt[bh][d][s]
      col[(size_t)(l15)      * S_] = (bf16)q0;
      col[(size_t)(l15 + 16) * S_] = (bf16)q1;
      col[(size_t)(l15 + 32) * S_] = (bf16)q2;
      col[(size_t)(l15 + 48) * S_] = (bf16)q3;
    }
  }
}

// ------------------------------- attention ---------------------------------
// grid (S/64, H, B), block 128 (4 waves sharing (b,h), 16 queries each).
// K/V chunks staged in LDS with double-buffered async global->LDS copies.
// Online softmax; P re-layout C->A via per-wave LDS tile.

#define KPAD (D_ + 8)   // 144B row stride, 16B aligned, bank-spread
#define VPAD (32 + 8)   // 80B row stride
#define PPAD (32 + 8)

__global__ __launch_bounds__(128) void attn_kernel(
    const bf16* __restrict__ Qb, const bf16* __restrict__ Kb,
    const bf16* __restrict__ Vt, bf16* __restrict__ attnb /* [B][S][E] */)
{
  __shared__ bf16 Kbuf[2][32][KPAD];
  __shared__ bf16 Vbuf[2][64][VPAD];
  __shared__ bf16 Pbuf[4][16][PPAD];

  const int tid   = threadIdx.x;
  const int lane  = tid & 31;
  const int wv    = tid >> 5;
  const int l15   = lane & 15;
  const int half  = lane >> 4;
  const int h     = blockIdx.y;
  const int b     = blockIdx.z;
  const int bh    = b * H_ + h;
  const int qbase = blockIdx.x * 64 + wv * 16;

  const bf16* Kg = Kb + (size_t)bh * S_ * D_;   // [key][dim]
  const bf16* Vg = Vt + (size_t)bh * D_ * S_;   // [dim][key]

  // Q tile (16 queries x 64) in A-layout registers, K = 64 -> two 16x32 slabs.
  const bf16* Qrow = Qb + ((size_t)bh * S_ + qbase + l15) * D_;
  V16 qa0, qa1;
  qa0.u[0] = *(const uint4*)(Qrow + 16 * half);
  qa0.u[1] = *(const uint4*)(Qrow + 16 * half + 8);
  qa1.u[0] = *(const uint4*)(Qrow + 32 + 16 * half);
  qa1.u[1] = *(const uint4*)(Qrow + 32 + 16 * half + 8);

  // Stage one 32-key chunk (K: 32x64, V^T: 64x32) into LDS buffer `bufi`.
  // 256+256 16B packets, 128 threads -> 2+2 packets per thread.
  auto stage = [&](int kc, int bufi) {
#pragma unroll
    for (int p = 0; p < 2; ++p) {
      const int c    = tid + p * 128;
      const int krow = c >> 2, kcol = (c & 3) * 8;   // K chunk packet
      async_copy16(Kg + (size_t)(kc + krow) * D_ + kcol, &Kbuf[bufi][krow][kcol]);
      const int vrow = c >> 1, vcol = (c & 1) * 8;   // V chunk packet
      async_copy16(Vg + (size_t)vrow * S_ + kc + vcol, &Vbuf[bufi][vrow][vcol]);
    }
  };

  v8f o0 = {}, o1 = {}, o2 = {}, o3 = {};
  float Mrow[8], Lrow[8];
#pragma unroll
  for (int r = 0; r < 8; ++r) { Mrow[r] = -1e30f; Lrow[r] = 0.0f; }

  stage(0, 0);

  for (int kc = 0; kc < S_; kc += 32) {
    const int bufi = (kc >> 5) & 1;
    async_wait_all();
    __syncthreads();                       // chunk `bufi` resident; prior reads done
    if (kc + 32 < S_) stage(kc + 32, bufi ^ 1);   // prefetch next chunk

    // ---- scores: S = Q * K^T (B-operand lane n = key, contiguous dims) ----
    v8f s0 = {}, s1 = {};
    {
      const bf16* kb0 = &Kbuf[bufi][l15][16 * half];
      const bf16* kb1 = &Kbuf[bufi][16 + l15][16 * half];
      V16 b0, b1;
      b0.u[0] = *(const uint4*)(kb0);      b0.u[1] = *(const uint4*)(kb0 + 8);
      b1.u[0] = *(const uint4*)(kb1);      b1.u[1] = *(const uint4*)(kb1 + 8);
      s0 = wmma_bf16(qa0.v, b0.v, s0);
      s1 = wmma_bf16(qa0.v, b1.v, s1);
      b0.u[0] = *(const uint4*)(kb0 + 32); b0.u[1] = *(const uint4*)(kb0 + 40);
      b1.u[0] = *(const uint4*)(kb1 + 32); b1.u[1] = *(const uint4*)(kb1 + 40);
      s0 = wmma_bf16(qa1.v, b0.v, s0);
      s1 = wmma_bf16(qa1.v, b1.v, s1);
    }

    // ---- online softmax (16-lane half-wave reductions match C layout) ----
    asm volatile("" ::: "memory");  // keep prior P reads before new P writes
    float corr8[8];
#pragma unroll
    for (int r = 0; r < 8; ++r) {
      float v0 = s0[r] * 0.125f;          // 1/sqrt(64)
      float v1 = s1[r] * 0.125f;
      float mx = fmaxf(v0, v1);
#pragma unroll
      for (int off = 1; off < 16; off <<= 1) mx = fmaxf(mx, __shfl_xor(mx, off, 32));
      const float Mn   = fmaxf(Mrow[r], mx);
      const float corr = __expf(Mrow[r] - Mn);
      const float p0   = __expf(v0 - Mn);
      const float p1   = __expf(v1 - Mn);
      float sum = p0 + p1;
#pragma unroll
      for (int off = 1; off < 16; off <<= 1) sum += __shfl_xor(sum, off, 32);
      Lrow[r]  = Lrow[r] * corr + sum;
      Mrow[r]  = Mn;
      corr8[r] = corr;
      const int m = r + 8 * half;
      Pbuf[wv][m][l15]      = (bf16)p0;
      Pbuf[wv][m][16 + l15] = (bf16)p1;
    }
#pragma unroll
    for (int r = 0; r < 8; ++r) {
      o0[r] *= corr8[r]; o1[r] *= corr8[r];
      o2[r] *= corr8[r]; o3[r] *= corr8[r];
    }

    // ---- re-layout P (C->A) via LDS; same-wave DS ops are in-order ----
    __builtin_amdgcn_wave_barrier();
    asm volatile("s_wait_dscnt 0" ::: "memory");
    V16 pa;
    const bf16* prow = &Pbuf[wv][l15][0];
    pa.u[0] = *(const uint4*)(prow + 16 * half);
    pa.u[1] = *(const uint4*)(prow + 16 * half + 8);

    // ---- O += P * V  (Vbuf rows contiguous along keys) ----
    {
      const bf16* vb0 = &Vbuf[bufi][l15][16 * half];
      const bf16* vb1 = &Vbuf[bufi][16 + l15][16 * half];
      const bf16* vb2 = &Vbuf[bufi][32 + l15][16 * half];
      const bf16* vb3 = &Vbuf[bufi][48 + l15][16 * half];
      V16 v0, v1, v2, v3;
      v0.u[0] = *(const uint4*)(vb0); v0.u[1] = *(const uint4*)(vb0 + 8);
      v1.u[0] = *(const uint4*)(vb1); v1.u[1] = *(const uint4*)(vb1 + 8);
      v2.u[0] = *(const uint4*)(vb2); v2.u[1] = *(const uint4*)(vb2 + 8);
      v3.u[0] = *(const uint4*)(vb3); v3.u[1] = *(const uint4*)(vb3 + 8);
      o0 = wmma_bf16(pa.v, v0.v, o0);
      o1 = wmma_bf16(pa.v, v1.v, o1);
      o2 = wmma_bf16(pa.v, v2.v, o2);
      o3 = wmma_bf16(pa.v, v3.v, o3);
    }
  }

  // ---- normalize and store to attnb [B][S][E] ----
#pragma unroll
  for (int r = 0; r < 8; ++r) {
    const int m    = r + 8 * half;
    const int sIdx = qbase + m;
    const float li = 1.0f / Lrow[r];
    bf16* row = attnb + ((size_t)(b * S_ + sIdx)) * E_ + h * D_;
    row[l15]      = (bf16)(o0[r] * li);
    row[l15 + 16] = (bf16)(o1[r] * li);
    row[l15 + 32] = (bf16)(o2[r] * li);
    row[l15 + 48] = (bf16)(o3[r] * li);
  }
}

// ---------------------------- output projection ----------------------------
// grid (NT/128, E/64), block 256 (8 waves). out = attn @ Wo + bo (fp32 out).

__global__ __launch_bounds__(256) void oproj_kernel(
    const bf16* __restrict__ attnb, const bf16* __restrict__ WoT,
    const float* __restrict__ bo, float* __restrict__ out)
{
  const int lane = threadIdx.x & 31;
  const int wv   = threadIdx.x >> 5;
  const int l15  = lane & 15;
  const int half = lane >> 4;
  const int m0   = blockIdx.x * 128 + wv * 16;
  const int n0   = blockIdx.y * 64;

  v8f acc0 = {}, acc1 = {}, acc2 = {}, acc3 = {};

  const bf16* Abase = attnb + (size_t)(m0 + l15) * E_;
  const bf16* B0    = WoT + (size_t)(n0 + l15)      * E_;
  const bf16* B1    = WoT + (size_t)(n0 + 16 + l15) * E_;
  const bf16* B2    = WoT + (size_t)(n0 + 32 + l15) * E_;
  const bf16* B3    = WoT + (size_t)(n0 + 48 + l15) * E_;

  for (int kk = 0; kk < E_; kk += 32) {
    V16 a;
    const bf16* ap = Abase + kk + 16 * half;
    a.u[0] = *(const uint4*)(ap);
    a.u[1] = *(const uint4*)(ap + 8);

    V16 b0, b1, b2, b3;
    const int bo2 = kk + 16 * half;
    b0.u[0] = *(const uint4*)(B0 + bo2); b0.u[1] = *(const uint4*)(B0 + bo2 + 8);
    b1.u[0] = *(const uint4*)(B1 + bo2); b1.u[1] = *(const uint4*)(B1 + bo2 + 8);
    b2.u[0] = *(const uint4*)(B2 + bo2); b2.u[1] = *(const uint4*)(B2 + bo2 + 8);
    b3.u[0] = *(const uint4*)(B3 + bo2); b3.u[1] = *(const uint4*)(B3 + bo2 + 8);

    acc0 = wmma_bf16(a.v, b0.v, acc0);
    acc1 = wmma_bf16(a.v, b1.v, acc1);
    acc2 = wmma_bf16(a.v, b2.v, acc2);
    acc3 = wmma_bf16(a.v, b3.v, acc3);
  }

  const float bias0 = bo[n0 + l15];
  const float bias1 = bo[n0 + 16 + l15];
  const float bias2 = bo[n0 + 32 + l15];
  const float bias3 = bo[n0 + 48 + l15];

#pragma unroll
  for (int r = 0; r < 8; ++r) {
    const int token = m0 + r + 8 * half;
    float* row = out + (size_t)token * E_ + n0;
    row[l15]      = acc0[r] + bias0;
    row[l15 + 16] = acc1[r] + bias1;
    row[l15 + 32] = acc2[r] + bias2;
    row[l15 + 48] = acc3[r] + bias3;
  }
}

// ------------------------------- launcher ----------------------------------

extern "C" void kernel_launch(void* const* d_in, const int* in_sizes, int n_in,
                              void* d_out, int out_size, void* d_ws, size_t ws_size,
                              hipStream_t stream) {
  const float* x  = (const float*)d_in[0];
  const float* Wq = (const float*)d_in[1];
  const float* bq = (const float*)d_in[2];
  const float* Wk = (const float*)d_in[3];
  const float* bk = (const float*)d_in[4];
  const float* Wv = (const float*)d_in[5];
  const float* bv = (const float*)d_in[6];
  const float* Wo = (const float*)d_in[7];
  const float* bo = (const float*)d_in[8];
  float* out = (float*)d_out;

  char* ws = (char*)d_ws;
  bf16* xb    = (bf16*)ws; ws += (size_t)NT * E_ * sizeof(bf16);   // 8 MB
  bf16* WqT   = (bf16*)ws; ws += (size_t)E_ * E_ * sizeof(bf16);   // 2 MB
  bf16* WkT   = (bf16*)ws; ws += (size_t)E_ * E_ * sizeof(bf16);
  bf16* WvT   = (bf16*)ws; ws += (size_t)E_ * E_ * sizeof(bf16);
  bf16* WoT   = (bf16*)ws; ws += (size_t)E_ * E_ * sizeof(bf16);
  bf16* Qb    = (bf16*)ws; ws += (size_t)NT * E_ * sizeof(bf16);   // [B][H][S][D]
  bf16* Kb    = (bf16*)ws; ws += (size_t)NT * E_ * sizeof(bf16);
  bf16* Vt    = (bf16*)ws; ws += (size_t)NT * E_ * sizeof(bf16);   // [B][H][D][S]
  bf16* attnb = (bf16*)ws; ws += (size_t)NT * E_ * sizeof(bf16);   // [B][S][E]

  // Phase 0: precision conversion + weight transpose
  cvt_x_kernel<<<(NT * E_) / 256, 256, 0, stream>>>(x, xb, NT * E_);
  cvt_wT_kernel<<<(E_ * E_) / 256, 256, 0, stream>>>(Wq, WqT);
  cvt_wT_kernel<<<(E_ * E_) / 256, 256, 0, stream>>>(Wk, WkT);
  cvt_wT_kernel<<<(E_ * E_) / 256, 256, 0, stream>>>(Wv, WvT);
  cvt_wT_kernel<<<(E_ * E_) / 256, 256, 0, stream>>>(Wo, WoT);

  // Phase 1: fused QKV projection + bias + RoPE (+ V transpose)
  dim3 g1(NT / 128, H_, 3);
  qkv_gemm_kernel<<<g1, 256, 0, stream>>>(xb, WqT, WkT, WvT, bq, bk, bv, Qb, Kb, Vt);

  // Phase 2: flash attention (async LDS staging, double-buffered)
  dim3 g2(S_ / 64, H_, B_);
  attn_kernel<<<g2, 128, 0, stream>>>(Qb, Kb, Vt, attnb);

  // Phase 3: output projection (fp32 result)
  dim3 g3(NT / 128, E_ / 64);
  oproj_kernel<<<g3, 256, 0, stream>>>(attnb, WoT, bo, out);
}
